// HypAgg_20366734917860
// MI455X (gfx1250) — matrix operand order repaired and checked by reading, not compiled, not run on hardware
//
#include <hip/hip_runtime.h>
#include <hip/hip_bf16.h>
#include <math.h>

typedef __attribute__((ext_vector_type(16))) _Float16 v16h;
typedef __attribute__((ext_vector_type(8)))  _Float16 v8h;
typedef __attribute__((ext_vector_type(8)))  float    v8f;

#define NPTS 1024
#define DDIM 64
#define KSPLIT 4

__device__ __forceinline__ float wave_sum(float v) {
#pragma unroll
  for (int o = 16; o > 0; o >>= 1) v += __shfl_xor(v, o, 32);
  return v;
}

__device__ __forceinline__ v16h cat8(v8h lo, v8h hi) {
  v16h r;
#pragma unroll
  for (int t = 0; t < 8; ++t) { r[t] = lo[t]; r[t + 8] = hi[t]; }
  return r;
}

// ---------------------------------------------------------------------------
// Kernel 1: per-row prep.  One wave (32 lanes) per row, 2 dims per lane.
//   x2[i] = ||x_i||^2
//   x_tan = logmap0(x);  left[i] = x_tan.W[:D] + b;  right[i] = x_tan.W[D:]
//   Xh  = fp16(x)  [N,D]      (GEMM A operands)
//   XhT = fp16(x)^T [D,N]     (GEMM2 B operand, contiguous-K)
// ---------------------------------------------------------------------------
__global__ void prep_kernel(const float* __restrict__ x,
                            const float* __restrict__ attW,
                            const float* __restrict__ attb,
                            _Float16* __restrict__ Xh,
                            _Float16* __restrict__ XhT,
                            float* __restrict__ x2o,
                            float* __restrict__ left,
                            float* __restrict__ right) {
  const int i  = blockIdx.x;
  const int t  = threadIdx.x;           // 0..31
  const int d0 = t, d1 = t + 32;
  const float v0 = x[i * DDIM + d0];
  const float v1 = x[i * DDIM + d1];

  const float x2 = wave_sum(v0 * v0 + v1 * v1);
  const float pn = fmaxf(sqrtf(x2), 1e-15f);
  const float z  = fminf(pn, 1.0f - 1e-7f);
  const float f  = atanhf(z) / pn;      // logmap0 scale (c = 1)
  const float xt0 = f * v0, xt1 = f * v1;

  const float ls = wave_sum(xt0 * attW[d0] + xt1 * attW[d1]);
  const float rs = wave_sum(xt0 * attW[DDIM + d0] + xt1 * attW[DDIM + d1]);
  if (t == 0) {
    x2o[i]   = x2;
    left[i]  = ls + attb[0];
    right[i] = rs;
  }
  Xh[i * DDIM + d0] = (_Float16)v0;
  Xh[i * DDIM + d1] = (_Float16)v1;
  XhT[d0 * NPTS + i] = (_Float16)v0;
  XhT[d1 * NPTS + i] = (_Float16)v1;
}

// ---------------------------------------------------------------------------
// Kernel 2: Gram matrix  G = X X^T   (f16 WMMA, f32 accumulate)
// One wave per 16x16 tile; 8 waves / block; 64x64 tiles.
// ---------------------------------------------------------------------------
__global__ void gram_kernel(const _Float16* __restrict__ Xh,
                            float* __restrict__ G) {
  const int lane   = threadIdx.x & 31;
  const int tileId = blockIdx.x * 8 + (threadIdx.x >> 5);   // 0..4095
  const int tileI  = tileId >> 6;
  const int tileJ  = tileId & 63;
  const int m      = lane & 15;
  const int hi     = lane >> 4;

  const _Float16* rowA = Xh + (tileI * 16 + m) * DDIM;
  const _Float16* rowB = Xh + (tileJ * 16 + m) * DDIM;

  v8f acc = {};
#pragma unroll
  for (int k0 = 0; k0 < DDIM; k0 += 32) {
    v8h alo = *(const v8h*)(rowA + k0 + hi * 8);
    v8h ahi = *(const v8h*)(rowA + k0 + 16 + hi * 8);
    v8h blo = *(const v8h*)(rowB + k0 + hi * 8);
    v8h bhi = *(const v8h*)(rowB + k0 + 16 + hi * 8);
    acc = __builtin_amdgcn_wmma_f32_16x16x32_f16(
        false, cat8(alo, ahi), false, cat8(blo, bhi),
        (short)0, acc, false, false);
  }
#pragma unroll
  for (int r = 0; r < 8; ++r)
    G[(tileI * 16 + r + hi * 8) * NPTS + tileJ * 16 + m] = acc[r];
}

// ---------------------------------------------------------------------------
// Kernel 3: pointwise [N,N] attention / logmap-scalar pass.
// One block per row i.  Emits Bh[i,j] (coef on x_j) and alpha[i] (coef on x_i).
// adj is ~2% dense and multiplies the whole weight -> skip the transcendental
// path (and the G load) entirely where adj == 0.
// ---------------------------------------------------------------------------
__global__ void att_kernel(const float* __restrict__ G,
                           const float* __restrict__ adj,
                           const float* __restrict__ x2v,
                           const float* __restrict__ left,
                           const float* __restrict__ right,
                           _Float16* __restrict__ Bh,
                           float* __restrict__ alpha) {
  const int i = blockIdx.x;
  const float x2i = x2v[i];
  const float li  = left[i];
  const float b   = 1.0f - x2i;                  // (1 - c*x2),  c = 1
  const float cpre = fmaxf(b, 1e-15f);           // 2/(sqrt_c*lambda_x)

  float asum = 0.0f;
  for (int j = threadIdx.x; j < NPTS; j += 256) {
    const float av = adj[i * NPTS + j];
    float wb = 0.0f;
    if (av != 0.0f) {
      const float g  = G[i * NPTS + j];          // x_i . x_j
      const float y2 = x2v[j];
      const float xy = -g;                       // (-x_i) . x_j
      const float a   = 1.0f + 2.0f * xy + y2;   // coef on (-x_i) side
      const float den = fmaxf(1.0f + 2.0f * xy + x2i * y2, 1e-15f);
      const float nn  = fmaxf(a * a * x2i + b * b * y2 - 2.0f * a * b * g, 0.0f);
      const float sn  = fmaxf(sqrtf(nn) / den, 1e-15f);
      const float z   = fminf(sn, 1.0f - 1e-7f);
      const float coef = cpre * atanhf(z) / sn;
      const float sig  = 1.0f / (1.0f + expf(-(li + right[j])));
      const float w    = sig * av * coef / den;
      wb = w * b;                                // weight on x_j
      asum += w * a;                             // weight on x_i (minus sign later)
    }
    Bh[i * NPTS + j] = (_Float16)wb;
  }
  __shared__ float red[8];
  const float ws = wave_sum(asum);
  if ((threadIdx.x & 31) == 0) red[threadIdx.x >> 5] = ws;
  __syncthreads();
  if (threadIdx.x == 0) {
    float s = 0.0f;
#pragma unroll
    for (int w = 0; w < 8; ++w) s += red[w];
    alpha[i] = s;
  }
}

// ---------------------------------------------------------------------------
// Kernel 4: support GEMM   Draw = B @ X   [1024x1024]x[1024x64]  (WMMA)
// Split-K by 4 for occupancy: 64(M) x 4(N) x 4(Ksplit) = 1024 waves,
// each accumulating a 256-deep K chunk into its own partial slab.
// ---------------------------------------------------------------------------
__global__ void support_kernel(const _Float16* __restrict__ Bh,
                               const _Float16* __restrict__ XhT,
                               float* __restrict__ Draw) {
  const int lane   = threadIdx.x & 31;
  const int tileId = blockIdx.x * 8 + (threadIdx.x >> 5);   // 0..1023
  const int ks     = tileId & 3;                            // K chunk
  const int tileN  = (tileId >> 2) & 3;
  const int tileM  = tileId >> 4;
  const int m      = lane & 15;
  const int hi     = lane >> 4;
  const int kbase  = ks * (NPTS / KSPLIT);

  const _Float16* rowA = Bh  + (tileM * 16 + m) * NPTS + kbase;
  const _Float16* rowB = XhT + (tileN * 16 + m) * NPTS + kbase;

  v8f acc = {};
  for (int k0 = 0; k0 < NPTS / KSPLIT; k0 += 32) {
    v8h alo = *(const v8h*)(rowA + k0 + hi * 8);
    v8h ahi = *(const v8h*)(rowA + k0 + 16 + hi * 8);
    v8h blo = *(const v8h*)(rowB + k0 + hi * 8);
    v8h bhi = *(const v8h*)(rowB + k0 + 16 + hi * 8);
    acc = __builtin_amdgcn_wmma_f32_16x16x32_f16(
        false, cat8(alo, ahi), false, cat8(blo, bhi),
        (short)0, acc, false, false);
  }
  float* slab = Draw + ks * (NPTS * DDIM);
#pragma unroll
  for (int r = 0; r < 8; ++r)
    slab[(tileM * 16 + r + hi * 8) * DDIM + tileN * 16 + m] = acc[r];
}

// ---------------------------------------------------------------------------
// Kernel 5: per-row  support_t = sum_k Draw_k - alpha*x ;
//           out = proj(expmap(support_t, x)).  One wave per row.
// ---------------------------------------------------------------------------
__global__ void expproj_kernel(const float* __restrict__ Draw,
                               const float* __restrict__ alpha,
                               const float* __restrict__ x,
                               const float* __restrict__ x2v,
                               float* __restrict__ out) {
  const int i = blockIdx.x;
  const int t = threadIdx.x;
  const int d0 = t, d1 = t + 32;
  const float al  = alpha[i];
  const float p0  = x[i * DDIM + d0], p1 = x[i * DDIM + d1];
  float u0 = -al * p0, u1 = -al * p1;
#pragma unroll
  for (int s = 0; s < KSPLIT; ++s) {
    u0 += Draw[s * (NPTS * DDIM) + i * DDIM + d0];
    u1 += Draw[s * (NPTS * DDIM) + i * DDIM + d1];
  }
  const float x2i = x2v[i];

  const float un2 = wave_sum(u0 * u0 + u1 * u1);
  const float pu  = wave_sum(p0 * u0 + p1 * u1);
  const float un  = fmaxf(sqrtf(un2), 1e-15f);
  const float lam = 2.0f / fmaxf(1.0f - x2i, 1e-15f);
  const float th  = tanhf(0.5f * lam * un);
  const float tt  = th / un;                 // second = tt * u
  const float s2  = tt * tt * un2;
  const float ps  = tt * pu;
  const float npc = 1.0f + 2.0f * ps + s2;   // coef on p
  const float nsc = 1.0f - x2i;              // coef on second
  const float den = fmaxf(1.0f + 2.0f * ps + x2i * s2, 1e-15f);
  float r0 = (npc * p0 + nsc * tt * u0) / den;
  float r1 = (npc * p1 + nsc * tt * u1) / den;

  const float rn = fmaxf(sqrtf(wave_sum(r0 * r0 + r1 * r1)), 1e-15f);
  const float maxn = 1.0f - 4e-3f;
  if (rn > maxn) { r0 *= maxn / rn; r1 *= maxn / rn; }
  out[i * DDIM + d0] = r0;
  out[i * DDIM + d1] = r1;
}

// ---------------------------------------------------------------------------
extern "C" void kernel_launch(void* const* d_in, const int* in_sizes, int n_in,
                              void* d_out, int out_size, void* d_ws, size_t ws_size,
                              hipStream_t stream) {
  const float* x    = (const float*)d_in[0];   // [1024,64]
  const float* adj  = (const float*)d_in[1];   // [1024,1024]
  const float* attW = (const float*)d_in[2];   // [128,1]
  const float* attb = (const float*)d_in[3];   // [1]
  float* out = (float*)d_out;

  char* ws = (char*)d_ws;
  _Float16* Xh   = (_Float16*)(ws + 0);                 // 128 KB
  _Float16* XhT  = (_Float16*)(ws + 131072);            // 128 KB
  float*    G    = (float*)   (ws + 262144);            // 4 MB
  _Float16* Bh   = (_Float16*)(ws + 4456448);           // 2 MB
  float*    Draw = (float*)   (ws + 6553600);           // 1 MB (4 partial slabs)
  float*    x2v  = (float*)   (ws + 7602176);           // 4 KB
  float*    left = (float*)   (ws + 7606272);
  float*    rght = (float*)   (ws + 7610368);
  float*    alph = (float*)   (ws + 7614464);

  prep_kernel<<<NPTS, 32, 0, stream>>>(x, attW, attb, Xh, XhT, x2v, left, rght);
  gram_kernel<<<512, 256, 0, stream>>>(Xh, G);
  att_kernel<<<NPTS, 256, 0, stream>>>(G, adj, x2v, left, rght, Bh, alph);
  support_kernel<<<128, 256, 0, stream>>>(Bh, XhT, Draw);
  expproj_kernel<<<NPTS, 32, 0, stream>>>(Draw, alph, x, x2v, out);
}
